// BigVGAN_18932215841050
// MI455X (gfx1250) — matrix operand order repaired and checked by reading, not compiled, not run on hardware
//
#include <hip/hip_runtime.h>
#include <math.h>

// ---------------------------------------------------------------------------
// BigVGAN forward for gfx1250 (MI455X).
// Convs = implicit GEMM on v_wmma_f32_16x16x32_bf16, f32 accumulate.
// 32(t) x 32(co) register tile per wave: 2 A frags x 2 B frags -> 4 WMMAs
// per 32-ci chunk. A staged via LDS transpose (bf16 packed-pair dwords),
// B read as b128s from per-layer repacked bf16 weights.
// ---------------------------------------------------------------------------

typedef __attribute__((ext_vector_type(16))) __bf16    v16bf;
typedef __attribute__((ext_vector_type(8)))  float     v8f;
typedef __attribute__((ext_vector_type(8)))  unsigned  v8u;

struct Filt12 { float f[12]; };

__device__ __forceinline__ int clampi(int v, int lo, int hi) {
    return v < lo ? lo : (v > hi ? hi : v);
}

// ---------------------------------------------------------------------------
// Weight repack: conv1d weights (Cout, Cin, KT) fp32 -> bf16 [KT][Coutp][Cinp]
// zero-padded so B-fragment loads are always in-bounds & branch-free.
// ---------------------------------------------------------------------------
__global__ void repack_conv(const float* __restrict__ w, __bf16* __restrict__ wb,
                            int Cout, int Cin, int KT, int Coutp, int Cinp)
{
    size_t idx = (size_t)blockIdx.x * blockDim.x + threadIdx.x;
    size_t n = (size_t)KT * Coutp * Cinp;
    if (idx >= n) return;
    int ci = (int)(idx % Cinp);
    int co = (int)((idx / Cinp) % Coutp);
    int q  = (int)(idx / ((size_t)Cinp * Coutp));
    float v = (co < Cout && ci < Cin) ? w[((size_t)co * Cin + ci) * KT + q] : 0.f;
    wb[idx] = (__bf16)v;
}

// ConvTranspose weights (Cin, Cout, K) fp32 -> bf16 [K][Coutp][Cinp]
__global__ void repack_convT(const float* __restrict__ w, __bf16* __restrict__ wb,
                             int Cin, int Cout, int K, int Coutp, int Cinp)
{
    size_t idx = (size_t)blockIdx.x * blockDim.x + threadIdx.x;
    size_t n = (size_t)K * Coutp * Cinp;
    if (idx >= n) return;
    int ci = (int)(idx % Cinp);
    int co = (int)((idx / Cinp) % Coutp);
    int kk = (int)(idx / ((size_t)Cinp * Coutp));
    float v = (co < Cout && ci < Cin) ? w[((size_t)ci * Cout + co) * K + kk] : 0.f;
    wb[idx] = (__bf16)v;
}

// A-fragment gather from LDS [t][ci] (bf16): row block at trow, packed pairs.
__device__ __forceinline__ v16bf gather_a(const unsigned* lds_u, int trow, int half)
{
    v8u au;
#pragma unroll
    for (int v = 0; v < 8; ++v) {
        int kbase = 2 * v + (half ? 16 : 0);
        au[v] = lds_u[trow * 16 + (kbase >> 1)];
    }
    return __builtin_bit_cast(v16bf, au);
}

// B-fragment: 8 consecutive dwords (16 bf16) from packed weight row.
__device__ __forceinline__ v16bf gather_b(const unsigned* wq, int half)
{
    v8u bu;
#pragma unroll
    for (int v = 0; v < 8; ++v) bu[v] = wq[v + (half ? 8 : 0)];
    return __builtin_bit_cast(v16bf, bu);
}

// ---------------------------------------------------------------------------
// Implicit-GEMM dilated conv1d (zero padding).
// One wave = 32(t) x 32(co) tile. K-loop: taps q (outer) x 32-ci chunks.
// grid: (L/32, Coutp/32, B), block: 32.
// ---------------------------------------------------------------------------
__global__ void conv1d_wmma(const float* __restrict__ x,
                            const __bf16* __restrict__ wb,  // [KT][Coutp][Cinp]
                            const float* __restrict__ bias,
                            const float* __restrict__ res,  // may be null
                            float* __restrict__ y,
                            int Cin, int Cout, int L, int KT, int dil, int pad,
                            int Coutp, int Cinp)
{
    __shared__ __bf16 lds_a[32 * 32];                // [t][ci], 2 KB
    const unsigned* lds_u = (const unsigned*)lds_a;

    const int lane = threadIdx.x & 31;
    const int t0   = blockIdx.x << 5;
    const int co0  = blockIdx.y << 5;
    const int b    = blockIdx.z;
    const int half = lane >> 4;
    const int nrow = lane & 15;

    const float* xb = x + (size_t)b * Cin * L;
    __builtin_prefetch(wb, 0, 1);

    v8f acc00 = {}, acc01 = {}, acc10 = {}, acc11 = {};
    for (int q = 0; q < KT; ++q) {
        const int tb = t0 + q * dil - pad;           // wave-uniform row base
        const bool interior = (tb >= 0) && (tb + 32 <= L);
        for (int c0 = 0; c0 < Cinp; c0 += 32) {
            // ---- stage A rows: lane owns channel c0+lane, 32 t values ----
            const int ci = c0 + lane;
            const float rowscale = (ci < Cin) ? 1.f : 0.f;
            const float* xr = xb + (size_t)clampi(ci, 0, Cin - 1) * L;
            float rv[32];
            if (interior) {                          // uniform branch, plain loads
#pragma unroll
                for (int t = 0; t < 32; ++t) rv[t] = xr[tb + t];
            } else {                                 // edge tile: clamp + mask
#pragma unroll
                for (int t = 0; t < 32; ++t) {
                    int tp = tb + t;
                    float m = (tp >= 0 && tp < L) ? 1.f : 0.f;
                    rv[t] = xr[clampi(tp, 0, L - 1)] * m;
                }
            }
#pragma unroll
            for (int t = 0; t < 32; ++t)
                lds_a[t * 32 + lane] = (__bf16)(rv[t] * rowscale);
            __syncthreads();

            v16bf A0 = gather_a(lds_u, nrow,      half);
            v16bf A1 = gather_a(lds_u, nrow + 16, half);

            const unsigned* wq0 = (const unsigned*)
                (wb + ((size_t)q * Coutp + co0 + nrow) * Cinp + c0);
            const unsigned* wq1 = wq0 + (size_t)16 * (Cinp >> 1);
            v16bf B0 = gather_b(wq0, half);
            v16bf B1 = gather_b(wq1, half);

            acc00 = __builtin_amdgcn_wmma_f32_16x16x32_bf16(
                false, A0, false, B0, (short)0, acc00, false, false);
            acc01 = __builtin_amdgcn_wmma_f32_16x16x32_bf16(
                false, A0, false, B1, (short)0, acc01, false, false);
            acc10 = __builtin_amdgcn_wmma_f32_16x16x32_bf16(
                false, A1, false, B0, (short)0, acc10, false, false);
            acc11 = __builtin_amdgcn_wmma_f32_16x16x32_bf16(
                false, A1, false, B1, (short)0, acc11, false, false);
            __syncthreads();                         // protect lds_a reuse
        }
    }

    // ---- store 4 sub-tiles: lane covers co = co0 + cj*16 + nrow,
    //      t block [t0 + ti*16 + half*8, +8) contiguous -> 2x float4 ----
#pragma unroll
    for (int cj = 0; cj < 2; ++cj) {
        const int co = co0 + cj * 16 + nrow;
        if (co >= Cout) continue;
        const float bsv = bias[co];
        const size_t row = ((size_t)b * Cout + co) * L;
        const v8f* accs[2] = { cj ? &acc01 : &acc00, cj ? &acc11 : &acc10 };
#pragma unroll
        for (int ti = 0; ti < 2; ++ti) {
            const v8f a = *accs[ti];
            const int tbase = t0 + ti * 16 + (half ? 8 : 0);
            float4 o0, o1;
            o0.x = a[0] + bsv; o0.y = a[1] + bsv; o0.z = a[2] + bsv; o0.w = a[3] + bsv;
            o1.x = a[4] + bsv; o1.y = a[5] + bsv; o1.z = a[6] + bsv; o1.w = a[7] + bsv;
            if (res) {
                const float4* rp = (const float4*)(res + row + tbase);
                float4 r0 = rp[0], r1 = rp[1];
                o0.x += r0.x; o0.y += r0.y; o0.z += r0.z; o0.w += r0.w;
                o1.x += r1.x; o1.y += r1.y; o1.z += r1.z; o1.w += r1.w;
            }
            float4* yp = (float4*)(y + row + tbase);
            yp[0] = o0; yp[1] = o1;
        }
    }
}

// ---------------------------------------------------------------------------
// Transposed conv (upsample), polyphase: phase p contributes taps
// kk = r + U*j (j in {0,1}), input index s = tt + s0off - j.
// 32(tt) x 32(co) tile per wave. grid: (Lin/32, Coutp/32, B*U), block: 32.
// ---------------------------------------------------------------------------
__global__ void convT_wmma(const float* __restrict__ x,
                           const __bf16* __restrict__ wb,  // [K][Coutp][Cinp]
                           const float* __restrict__ bias,
                           float* __restrict__ y,
                           int Cin, int Cout, int Lin, int U, int K,
                           int Coutp, int Cinp)
{
    __shared__ __bf16 lds_a[32 * 32];
    const unsigned* lds_u = (const unsigned*)lds_a;

    const int lane  = threadIdx.x & 31;
    const int tt0   = blockIdx.x << 5;
    const int co0   = blockIdx.y << 5;
    const int phase = blockIdx.z % U;
    const int b     = blockIdx.z / U;
    const int half  = lane >> 4;
    const int nrow  = lane & 15;

    const int pad   = (K - U) / 2;
    const int r     = (phase + pad) % U;
    const int s0off = (phase + pad - r) / U;         // 0 or 1

    const float* xb = x + (size_t)b * Cin * Lin;
    __builtin_prefetch(wb, 0, 1);

    v8f acc00 = {}, acc01 = {}, acc10 = {}, acc11 = {};
#pragma unroll
    for (int j = 0; j < 2; ++j) {
        const int tb = tt0 + s0off - j;              // wave-uniform
        const bool interior = (tb >= 0) && (tb + 32 <= Lin);
        const int kk = r + U * j;
        for (int c0 = 0; c0 < Cinp; c0 += 32) {
            const int ci = c0 + lane;
            const float rowscale = (ci < Cin) ? 1.f : 0.f;
            const float* xr = xb + (size_t)clampi(ci, 0, Cin - 1) * Lin;
            float rv[32];
            if (interior) {
#pragma unroll
                for (int t = 0; t < 32; ++t) rv[t] = xr[tb + t];
            } else {
#pragma unroll
                for (int t = 0; t < 32; ++t) {
                    int tp = tb + t;
                    float m = (tp >= 0 && tp < Lin) ? 1.f : 0.f;
                    rv[t] = xr[clampi(tp, 0, Lin - 1)] * m;
                }
            }
#pragma unroll
            for (int t = 0; t < 32; ++t)
                lds_a[t * 32 + lane] = (__bf16)(rv[t] * rowscale);
            __syncthreads();

            v16bf A0 = gather_a(lds_u, nrow,      half);
            v16bf A1 = gather_a(lds_u, nrow + 16, half);

            const unsigned* wq0 = (const unsigned*)
                (wb + ((size_t)kk * Coutp + co0 + nrow) * Cinp + c0);
            const unsigned* wq1 = wq0 + (size_t)16 * (Cinp >> 1);
            v16bf B0 = gather_b(wq0, half);
            v16bf B1 = gather_b(wq1, half);

            acc00 = __builtin_amdgcn_wmma_f32_16x16x32_bf16(
                false, A0, false, B0, (short)0, acc00, false, false);
            acc01 = __builtin_amdgcn_wmma_f32_16x16x32_bf16(
                false, A0, false, B1, (short)0, acc01, false, false);
            acc10 = __builtin_amdgcn_wmma_f32_16x16x32_bf16(
                false, A1, false, B0, (short)0, acc10, false, false);
            acc11 = __builtin_amdgcn_wmma_f32_16x16x32_bf16(
                false, A1, false, B1, (short)0, acc11, false, false);
            __syncthreads();
        }
    }

    const int Lout = Lin * U;
#pragma unroll
    for (int cj = 0; cj < 2; ++cj) {
        const int co = co0 + cj * 16 + nrow;
        if (co >= Cout) continue;
        const float bsv = bias[co];
        float* yb = y + ((size_t)b * Cout + co) * Lout;
        const v8f* accs[2] = { cj ? &acc01 : &acc00, cj ? &acc11 : &acc10 };
#pragma unroll
        for (int ti = 0; ti < 2; ++ti) {
            const v8f a = *accs[ti];
#pragma unroll
            for (int rr = 0; rr < 8; ++rr) {
                int tt = tt0 + ti * 16 + rr + (half ? 8 : 0);
                yb[(size_t)tt * U + phase] = a[rr] + bsv;
            }
        }
    }
}

// ---------------------------------------------------------------------------
// Fused 2x kaiser-sinc upsample (edge pad, transposed FIR, crop) + snakebeta.
//   t even:  u = 2*sum_{j=0..5} F[11-2j] * x[clamp(t/2 + j - 3)]
//   t odd :  u = 2*sum_{j=0..5} F[10-2j] * x[clamp((t-1)/2 + j - 2)]
//   out = u + (1/(exp(beta_c)+1e-9)) * sin(exp(alpha_c)*u)^2
// ---------------------------------------------------------------------------
__global__ void up2_snake(const float* __restrict__ xin,
                          float* __restrict__ out,
                          const float* __restrict__ alpha,
                          const float* __restrict__ beta,
                          int C, int L, int B, Filt12 F)
{
    const int L2 = 2 * L;
    size_t idx = (size_t)blockIdx.x * blockDim.x + threadIdx.x;
    size_t total = (size_t)B * C * L2;
    if (idx >= total) return;
    int t = (int)(idx % L2);
    int c = (int)((idx / L2) % C);
    size_t bc = idx / (size_t)L2;
    const float* xr = xin + bc * L;

    float acc = 0.f;
    if (t & 1) {
        int base = ((t - 1) >> 1) - 2;
#pragma unroll
        for (int j = 0; j < 6; ++j)
            acc += F.f[10 - 2 * j] * xr[clampi(base + j, 0, L - 1)];
    } else {
        int base = (t >> 1) - 3;
#pragma unroll
        for (int j = 0; j < 6; ++j)
            acc += F.f[11 - 2 * j] * xr[clampi(base + j, 0, L - 1)];
    }
    float u  = 2.f * acc;
    float ea = __expf(alpha[c]);
    float eb = __expf(beta[c]);
    float sn = __sinf(ea * u);
    out[idx] = u + (1.f / (eb + 1e-9f)) * sn * sn;
}

// 2x downsample: replicate pad (5,6), stride-2 FIR.
__global__ void down2_fir(const float* __restrict__ sin_,
                          float* __restrict__ out,
                          int C, int L, int B, Filt12 F)
{
    size_t idx = (size_t)blockIdx.x * blockDim.x + threadIdx.x;
    size_t total = (size_t)B * C * L;
    if (idx >= total) return;
    int t = (int)(idx % L);
    size_t bc = idx / (size_t)L;
    const int L2 = 2 * L;
    const float* sr = sin_ + bc * L2;
    float acc = 0.f;
#pragma unroll
    for (int k = 0; k < 12; ++k)
        acc += F.f[k] * sr[clampi(2 * t + k - 5, 0, L2 - 1)];
    out[idx] = acc;
}

__global__ void zero_kernel(float* __restrict__ p, size_t n) {
    size_t i = (size_t)blockIdx.x * blockDim.x + threadIdx.x;
    if (i < n) p[i] = 0.f;
}
__global__ void add_into(float* __restrict__ dst, const float* __restrict__ src, size_t n) {
    size_t i = (size_t)blockIdx.x * blockDim.x + threadIdx.x;
    if (i < n) dst[i] += src[i];
}
__global__ void scale_to(float* __restrict__ dst, const float* __restrict__ src,
                         float s, size_t n) {
    size_t i = (size_t)blockIdx.x * blockDim.x + threadIdx.x;
    if (i < n) dst[i] = src[i] * s;
}

// conv_post (Cin=8, K=7, pad=3) fused with tanh. out (B,1,L). Branchless taps.
__global__ void conv_post_tanh(const float* __restrict__ x,
                               const float* __restrict__ w,   // (1,8,7)
                               const float* __restrict__ bias,
                               float* __restrict__ out, int L, int B)
{
    size_t idx = (size_t)blockIdx.x * blockDim.x + threadIdx.x;
    if (idx >= (size_t)B * L) return;
    int t = (int)(idx % L);
    int b = (int)(idx / L);
    const float* xb = x + (size_t)b * 8 * L;
    float acc = bias[0];
#pragma unroll
    for (int q = 0; q < 7; ++q) {
        int tp = t + q - 3;
        float m = (tp >= 0 && tp < L) ? 1.f : 0.f;
        int tc = clampi(tp, 0, L - 1);
#pragma unroll
        for (int ci = 0; ci < 8; ++ci)
            acc += w[ci * 7 + q] * xb[(size_t)ci * L + tc] * m;
    }
    out[idx] = tanhf(acc);
}

// ---------------------------------------------------------------------------
// Host side
// ---------------------------------------------------------------------------
static void make_filter(Filt12& F)
{
    const double cutoff = 0.25, half_width = 0.3;
    const int K = 12, half = 6;
    double delta_f = 4.0 * half_width;
    double A = 2.285 * (half - 1) * M_PI * delta_f + 7.95;
    double beta;
    if (A > 50.0)      beta = 0.1102 * (A - 8.7);
    else if (A >= 21.) beta = 0.5842 * pow(A - 21.0, 0.4) + 0.07886 * (A - 21.0);
    else               beta = 0.0;
    auto i0 = [](double x) {
        double s = 1.0, t = 1.0;
        for (int k = 1; k < 40; ++k) { double u = x / (2.0 * k); t *= u * u; s += t; }
        return s;
    };
    double fv[12], sum = 0.0;
    for (int n = 0; n < K; ++n) {
        double r = (n - (K - 1) / 2.0) / ((K - 1) / 2.0);
        double wn = i0(beta * sqrt(fmax(0.0, 1.0 - r * r))) / i0(beta);
        double tm = (n - half) + 0.5;
        double xx = 2.0 * cutoff * tm;
        double sinc = (xx == 0.0) ? 1.0 : sin(M_PI * xx) / (M_PI * xx);
        fv[n] = 2.0 * cutoff * wn * sinc;
        sum += fv[n];
    }
    for (int n = 0; n < K; ++n) F.f[n] = (float)(fv[n] / sum);
}

static inline dim3 ew_grid(size_t n) { return dim3((unsigned)((n + 255) / 256)); }
static inline int rup(int v, int m) { return (v + m - 1) & ~(m - 1); }

extern "C" void kernel_launch(void* const* d_in, const int* in_sizes, int n_in,
                              void* d_out, int out_size, void* d_ws, size_t ws_size,
                              hipStream_t stream)
{
    (void)in_sizes; (void)n_in; (void)out_size; (void)ws_size;

    const int B = 2;
    Filt12 F; make_filter(F);

    auto in = [&](int i) { return (const float*)d_in[i]; };
    // Flattening order: 0:x | 1,2:conv_pre w,b | 3..14:ups[i].{w,b} |
    // 15+6j: resblocks[j].{c1w,c1b,c2w,c2b,alpha,beta} | 123,124:act_post |
    // 125,126: conv_post w,b
    const float* X0      = in(0);
    const float* cprw    = in(1);
    const float* cprb    = in(2);
    const float* apost_a = in(123);
    const float* apost_b = in(124);
    const float* cpw     = in(125);
    const float* cpb     = in(126);

    float* ws = (float*)d_ws;
    const size_t SM = 2097152;            // 8 MB slots (floats)
    float* P  = ws + 0 * SM;              // stage activation
    float* Q  = ws + 1 * SM;              // upsample-conv output
    float* XS = ws + 2 * SM;              // resblock accumulator
    float* R  = ws + 3 * SM;              // running x inside AMP block
    float* T  = ws + 4 * SM;              // post-downsample
    float* V  = ws + 5 * SM;              // conv1 output
    float* U  = ws + 6 * SM;              // 2x-rate slot (2 SM wide)
    __bf16* WB = (__bf16*)(ws + 8 * SM);  // packed-weight slot (2 MB)

    const int rates[6] = {4, 4, 2, 2, 2, 2};
    const int upks[6]  = {8, 8, 4, 4, 4, 4};
    const int rbk[3]   = {3, 7, 11};
    const int dls[3]   = {1, 3, 5};

    auto run_conv = [&](const float* xin, const float* w, const float* bias,
                        const float* res, float* yout,
                        int Cin, int Cout, int L, int KT, int dil, int pad) {
        const int Cinp = rup(Cin, 32), Coutp = rup(Cout, 32);
        const size_t nw = (size_t)KT * Coutp * Cinp;
        repack_conv<<<ew_grid(nw), 256, 0, stream>>>(w, WB, Cout, Cin, KT, Coutp, Cinp);
        dim3 g(L / 32, Coutp / 32, B);
        conv1d_wmma<<<g, 32, 0, stream>>>(xin, WB, bias, res, yout,
                                          Cin, Cout, L, KT, dil, pad, Coutp, Cinp);
    };

    // ---- conv_pre: (B,80,512) -> P (B,512,512) ----
    run_conv(X0, cprw, cprb, nullptr, P, 80, 512, 512, 7, 1, 3);

    int C = 512, L = 512;
    for (int i = 0; i < 6; ++i) {
        const int u = rates[i], K = upks[i];
        const int Cout = C / 2, Lout = L * u;
        const float* uw = in(3 + 2 * i);
        const float* ub = in(3 + 2 * i + 1);

        // transposed conv: P (B,C,L) -> Q (B,Cout,Lout)
        {
            const int Cinp = rup(C, 32), Coutp = rup(Cout, 32);
            const size_t nw = (size_t)K * Coutp * Cinp;
            repack_convT<<<ew_grid(nw), 256, 0, stream>>>(uw, WB, C, Cout, K, Coutp, Cinp);
            dim3 g(L / 32, Coutp / 32, B * u);
            convT_wmma<<<g, 32, 0, stream>>>(P, WB, ub, Q, C, Cout, L, u, K, Coutp, Cinp);
        }

        const size_t nact = (size_t)B * Cout * Lout;
        zero_kernel<<<ew_grid(nact), 256, 0, stream>>>(XS, nact);

        for (int j = 0; j < 3; ++j) {
            const int rbi  = i * 3 + j;
            const int base = 15 + 6 * rbi;
            const float* c1w = in(base + 0);
            const float* c1b = in(base + 1);
            const float* c2w = in(base + 2);
            const float* c2b = in(base + 3);
            const float* al  = in(base + 4);   // (6, Cout)
            const float* be  = in(base + 5);
            const int kk = rbk[j];

            hipMemcpyAsync(R, Q, nact * sizeof(float),
                           hipMemcpyDeviceToDevice, stream);

            for (int d = 0; d < 3; ++d) {
                const int dd = dls[d];
                // aa_act(R) -> T
                up2_snake<<<ew_grid(nact * 2), 256, 0, stream>>>(
                    R, U, al + (size_t)(2 * d) * Cout, be + (size_t)(2 * d) * Cout,
                    Cout, Lout, B, F);
                down2_fir<<<ew_grid(nact), 256, 0, stream>>>(U, T, Cout, Lout, B, F);
                // conv1 (dilated) -> V
                run_conv(T, c1w + (size_t)d * Cout * Cout * kk,
                         c1b + (size_t)d * Cout, nullptr, V,
                         Cout, Cout, Lout, kk, dd, dd * (kk - 1) / 2);
                // aa_act(V) -> T
                up2_snake<<<ew_grid(nact * 2), 256, 0, stream>>>(
                    V, U, al + (size_t)(2 * d + 1) * Cout, be + (size_t)(2 * d + 1) * Cout,
                    Cout, Lout, B, F);
                down2_fir<<<ew_grid(nact), 256, 0, stream>>>(U, T, Cout, Lout, B, F);
                // conv2 + residual (R updated in place; res read is same-element)
                run_conv(T, c2w + (size_t)d * Cout * Cout * kk,
                         c2b + (size_t)d * Cout, R, R,
                         Cout, Cout, Lout, kk, 1, (kk - 1) / 2);
            }
            add_into<<<ew_grid(nact), 256, 0, stream>>>(XS, R, nact);
        }
        scale_to<<<ew_grid(nact), 256, 0, stream>>>(P, XS, 1.f / 3.f, nact);
        C = Cout; L = Lout;
    }

    // ---- act_post + conv_post + tanh ----
    {
        const size_t nact = (size_t)B * C * L;      // C == 8, L == 131072
        up2_snake<<<ew_grid(nact * 2), 256, 0, stream>>>(P, U, apost_a, apost_b,
                                                         C, L, B, F);
        down2_fir<<<ew_grid(nact), 256, 0, stream>>>(U, T, C, L, B, F);
        conv_post_tanh<<<ew_grid((size_t)B * L), 256, 0, stream>>>(
            T, cpw, cpb, (float*)d_out, L, B);
    }
}
